// MultiHeadAttention_15264313770408
// MI455X (gfx1250) — matrix operand order repaired and checked by reading, not compiled, not run on hardware
//
#include <hip/hip_runtime.h>
#include <hip/hip_bf16.h>

typedef __attribute__((ext_vector_type(16))) __bf16 v16bf;
typedef __attribute__((ext_vector_type(8)))  __bf16 v8bf;
typedef __attribute__((ext_vector_type(8)))  float  v8f;

#define B_   8
#define H_   8
#define S_   1024
#define D_   512
#define DH_  64
#define NTOK_ (B_ * S_)
#define NEGV (-1e9f)

static __device__ __forceinline__ v16bf cat16(v8bf lo, v8bf hi) {
  return __builtin_shufflevector(lo, hi, 0,1,2,3,4,5,6,7,8,9,10,11,12,13,14,15);
}
static __device__ __forceinline__ v8bf ld8(const __bf16* p) {
  return *(const v8bf*)p;
}

// ---------------------------------------------------------------- f32 -> bf16
__global__ __launch_bounds__(256)
void mha_cvt_bf16(const float* __restrict__ src, __bf16* __restrict__ dst, int n8) {
  int i = blockIdx.x * blockDim.x + threadIdx.x;
  if (i >= n8) return;
  const float4* s4 = (const float4*)src;
  float4 f0 = s4[2*i+0], f1 = s4[2*i+1];
  v8bf o;
  o[0]=(__bf16)f0.x; o[1]=(__bf16)f0.y; o[2]=(__bf16)f0.z; o[3]=(__bf16)f0.w;
  o[4]=(__bf16)f1.x; o[5]=(__bf16)f1.y; o[6]=(__bf16)f1.z; o[7]=(__bf16)f1.w;
  ((v8bf*)dst)[i] = o;
}

// ------------------------------------------- relu(X @ W^T + bias), head split
// One wave computes a 16-row x 64-col (one full head) output tile: the A
// fragment is loaded once per K-step and reused by 4 WMMAs.
// vt==0: Out[(b*H+h)*S + s][dd]  (row-major per head, for Q and K)
// vt==1: Out[(b*H+h)*DH + dd][s] (transposed per head, for V)
__global__ __launch_bounds__(32)
void mha_qkv_proj(const __bf16* __restrict__ X, const __bf16* __restrict__ W,
                  const float* __restrict__ bias, __bf16* __restrict__ Out, int vt) {
  const int lane = threadIdx.x;
  const int half = lane >> 4, idx = lane & 15;
  const int h  = blockIdx.x;          // head -> columns [h*64, h*64+64)
  const int i0 = blockIdx.y * 16;     // output row tile
  const __bf16* arow = X + (size_t)(i0 + idx) * D_;

  v8f acc[4];
  {
    v8f z = {};
#pragma unroll
    for (int t = 0; t < 4; ++t) acc[t] = z;
  }

#pragma unroll 2
  for (int k0 = 0; k0 < D_; k0 += 32) {
    // A layout: lanes 0-15 K={k0..k0+7, k0+16..k0+23}, lanes 16-31 shifted by 8
    v16bf a = cat16(ld8(arow + k0 + half*8), ld8(arow + k0 + 16 + half*8));
#pragma unroll
    for (int t = 0; t < 4; ++t) {
      const __bf16* brow = W + (size_t)(h*DH_ + t*16 + idx) * D_;
      // B layout: lanes 0-15 K={k0..k0+7, k0+16..k0+23}, lanes 16-31 shifted by 8
      v16bf wb = cat16(ld8(brow + k0 + half*8), ld8(brow + k0 + 16 + half*8));
      acc[t] = __builtin_amdgcn_wmma_f32_16x16x32_bf16(false, a, false, wb,
                                                       (short)0, acc[t], false, false);
    }
  }

#pragma unroll
  for (int t = 0; t < 4; ++t) {
    const int dd = t*16 + idx;
    const float bj = bias[h*DH_ + dd];
#pragma unroll
    for (int r = 0; r < 8; ++r) {
      int row = i0 + r + half*8;
      int bb = row >> 10, s = row & (S_ - 1);
      float v = acc[t][r] + bj;
      v = v > 0.f ? v : 0.f;
      size_t o = vt ? (((size_t)(bb*H_ + h) * DH_ + dd) * S_ + s)
                    : (((size_t)(bb*H_ + h) * S_  + s) * DH_ + dd);
      Out[o] = (__bf16)v;
    }
  }
}

// ------------------------------------------------ flash attention, one wave per
// (batch-head, 32 query rows). Qh/Kh: [BH,S,64] bf16; Vt: [BH,64,S] bf16.
// K/V fragments are loaded once per 32-key tile and reused by both 16-query
// groups. NOTE reference mask quirk: mask row for batch-head i=b*H+h is i%B.
__global__ __launch_bounds__(32)
void mha_attention(const __bf16* __restrict__ Qh, const __bf16* __restrict__ Kh,
                   const __bf16* __restrict__ Vt, const int* __restrict__ qmask,
                   const int* __restrict__ kmask, float* __restrict__ Y) {
  __shared__ __bf16 Plds[2 * 16 * 32];
  const int lane = threadIdx.x;
  const int half = lane >> 4, idx = lane & 15;
  const int bh = blockIdx.y;
  const int b  = bh >> 3, h = bh & 7;
  const int mb = bh % B_;             // mask row index (== h here)
  const int q0 = blockIdx.x * 32;
  const __bf16* Qb = Qh + (size_t)bh * S_ * DH_;
  const __bf16* Kb = Kh + (size_t)bh * S_ * DH_;
  const __bf16* Vb = Vt + (size_t)bh * DH_ * S_;

  v16bf aq[2][2];
#pragma unroll
  for (int g = 0; g < 2; ++g) {
    const __bf16* qrow = Qb + (size_t)(q0 + g*16 + idx) * DH_;
    aq[g][0] = cat16(ld8(qrow + half*8),      ld8(qrow + 16 + half*8));
    aq[g][1] = cat16(ld8(qrow + 32 + half*8), ld8(qrow + 48 + half*8));
  }

  v8f O[2][4];
  float mrow[2][8], lrow[2][8];
  {
    v8f z = {};
#pragma unroll
    for (int g = 0; g < 2; ++g) {
#pragma unroll
      for (int c = 0; c < 4; ++c) O[g][c] = z;
#pragma unroll
      for (int r = 0; r < 8; ++r) { mrow[g][r] = -__builtin_inff(); lrow[g][r] = 0.f; }
    }
  }
  const int* km = kmask + (size_t)mb * S_;

  for (int kt = 0; kt < S_; kt += 32) {
    if (kt + 32 < S_)
      __builtin_prefetch(Kb + (size_t)(kt + 32 + idx) * DH_, 0, 1);

    // ---- scores: 2 key subtiles x 2 query groups, K/V frags loaded once
    v8f sc[2][2];   // [g][t]
    {
      v8f z = {};
#pragma unroll
      for (int g = 0; g < 2; ++g) { sc[g][0] = z; sc[g][1] = z; }
    }
#pragma unroll
    for (int t = 0; t < 2; ++t) {
      const __bf16* krow = Kb + (size_t)(kt + t*16 + idx) * DH_;
      v16bf b0 = cat16(ld8(krow + half*8),      ld8(krow + 16 + half*8));
      v16bf b1 = cat16(ld8(krow + 32 + half*8), ld8(krow + 48 + half*8));
#pragma unroll
      for (int g = 0; g < 2; ++g) {
        sc[g][t] = __builtin_amdgcn_wmma_f32_16x16x32_bf16(false, aq[g][0], false, b0,
                                                           (short)0, sc[g][t], false, false);
        sc[g][t] = __builtin_amdgcn_wmma_f32_16x16x32_bf16(false, aq[g][1], false, b1,
                                                           (short)0, sc[g][t], false, false);
      }
    }
    const int km0 = km[kt + idx], km1 = km[kt + 16 + idx];

    // ---- online softmax + C-layout -> A-layout transpose through LDS
#pragma unroll
    for (int g = 0; g < 2; ++g) {
#pragma unroll
      for (int r = 0; r < 8; ++r) {
        float v0 = km0 ? sc[g][0][r] * 0.125f : NEGV;
        float v1 = km1 ? sc[g][1][r] * 0.125f : NEGV;
        float t = fmaxf(v0, v1);                     // row max over 16 lanes
        t = fmaxf(t, __shfl_xor(t, 1));
        t = fmaxf(t, __shfl_xor(t, 2));
        t = fmaxf(t, __shfl_xor(t, 4));
        t = fmaxf(t, __shfl_xor(t, 8));
        float mn    = fmaxf(mrow[g][r], t);
        float alpha = __expf(mrow[g][r] - mn);
        float p0 = __expf(v0 - mn);
        float p1 = __expf(v1 - mn);
        float rs = p0 + p1;
        rs += __shfl_xor(rs, 1);
        rs += __shfl_xor(rs, 2);
        rs += __shfl_xor(rs, 4);
        rs += __shfl_xor(rs, 8);
        lrow[g][r] = lrow[g][r] * alpha + rs;
        mrow[g][r] = mn;
        O[g][0][r] *= alpha; O[g][1][r] *= alpha;
        O[g][2][r] *= alpha; O[g][3][r] *= alpha;
        const int Mr = r + half*8;
        Plds[g*512 + Mr*32 + idx]      = (__bf16)p0;
        Plds[g*512 + Mr*32 + 16 + idx] = (__bf16)p1;
      }
    }
    asm volatile("s_wait_dscnt 0" ::: "memory");

    v16bf pa[2];
#pragma unroll
    for (int g = 0; g < 2; ++g) {
      const __bf16* prow = &Plds[g*512 + idx * 32];  // read back in A layout
      pa[g] = cat16(ld8(prow + half*8), ld8(prow + 16 + half*8));
    }

    // ---- PV: V fragment loaded once per d-chunk, reused by both query groups
#pragma unroll
    for (int c = 0; c < 4; ++c) {
      const __bf16* vr = Vb + (size_t)(c*16 + idx) * S_ + kt + half*8;
      v16bf vb = cat16(ld8(vr), ld8(vr + 16));
#pragma unroll
      for (int g = 0; g < 2; ++g)
        O[g][c] = __builtin_amdgcn_wmma_f32_16x16x32_bf16(false, pa[g], false, vb,
                                                          (short)0, O[g][c], false, false);
    }
    asm volatile("s_wait_dscnt 0" ::: "memory");
  }

  const int* qm = qmask + (size_t)mb * S_;
#pragma unroll
  for (int g = 0; g < 2; ++g) {
#pragma unroll
    for (int r = 0; r < 8; ++r) {
      int s = q0 + g*16 + r + half*8;
      float f = (qm[s] != 0 ? 1.f : 0.f) / lrow[g][r];
      float* yp = Y + ((size_t)(b * S_ + s) * D_) + h * DH_;
      yp[idx]      = O[g][0][r] * f;
      yp[16 + idx] = O[g][1][r] * f;
      yp[32 + idx] = O[g][2][r] * f;
      yp[48 + idx] = O[g][3][r] * f;
    }
  }
}

// ---------------------------------------------- residual + LayerNorm, 1 wave/row
__global__ __launch_bounds__(32)
void mha_ln(const float* __restrict__ Y, const float* __restrict__ Qin,
            const float* __restrict__ gamma, const float* __restrict__ beta,
            float* __restrict__ out) {
  const int row  = blockIdx.x;
  const int lane = threadIdx.x;
  const size_t base = (size_t)row * D_ + lane * 16;
  float v[16];
  const float4* y4 = (const float4*)(Y + base);
  const float4* q4 = (const float4*)(Qin + base);
#pragma unroll
  for (int i = 0; i < 4; ++i) {
    float4 a = y4[i], qv = q4[i];
    v[4*i+0] = a.x + qv.x; v[4*i+1] = a.y + qv.y;
    v[4*i+2] = a.z + qv.z; v[4*i+3] = a.w + qv.w;
  }
  float s = 0.f, s2 = 0.f;
#pragma unroll
  for (int i = 0; i < 16; ++i) { s += v[i]; s2 += v[i]*v[i]; }
  s  += __shfl_xor(s, 1);  s2 += __shfl_xor(s2, 1);
  s  += __shfl_xor(s, 2);  s2 += __shfl_xor(s2, 2);
  s  += __shfl_xor(s, 4);  s2 += __shfl_xor(s2, 4);
  s  += __shfl_xor(s, 8);  s2 += __shfl_xor(s2, 8);
  s  += __shfl_xor(s, 16); s2 += __shfl_xor(s2, 16);
  float mean = s  * (1.f / D_);
  float var  = s2 * (1.f / D_) - mean * mean;
  float rstd = rsqrtf(var + 1e-6f);
  const float* g  = gamma + lane * 16;
  const float* bt = beta  + lane * 16;
  float* o = out + base;
#pragma unroll
  for (int i = 0; i < 16; ++i)
    o[i] = g[i] * (v[i] - mean) * rstd + bt[i];
}

extern "C" void kernel_launch(void* const* d_in, const int* in_sizes, int n_in,
                              void* d_out, int out_size, void* d_ws, size_t ws_size,
                              hipStream_t stream) {
  (void)in_sizes; (void)n_in; (void)out_size; (void)ws_size;
  const float* q     = (const float*)d_in[0];
  const float* k     = (const float*)d_in[1];
  const float* v     = (const float*)d_in[2];
  const int*   qmask = (const int*)d_in[3];
  const int*   kmask = (const int*)d_in[4];
  const float* Wq    = (const float*)d_in[5];
  const float* bq    = (const float*)d_in[6];
  const float* Wk    = (const float*)d_in[7];
  const float* bk    = (const float*)d_in[8];
  const float* Wv    = (const float*)d_in[9];
  const float* bv    = (const float*)d_in[10];
  const float* gamma = (const float*)d_in[11];
  const float* beta  = (const float*)d_in[12];
  float* out = (float*)d_out;

  char* ws = (char*)d_ws;
  size_t off = 0;
  auto carve = [&](size_t bytes) -> char* {
    char* p = ws + off;
    off += (bytes + 255) & ~(size_t)255;
    return p;
  };
  const size_t actN = (size_t)NTOK_ * D_;   // 4,194,304 elems
  const size_t wN   = (size_t)D_ * D_;      //   262,144 elems
  __bf16* qbf  = (__bf16*)carve(actN * 2);
  __bf16* kbf  = (__bf16*)carve(actN * 2);
  __bf16* vbf  = (__bf16*)carve(actN * 2);
  __bf16* wqbf = (__bf16*)carve(wN * 2);
  __bf16* wkbf = (__bf16*)carve(wN * 2);
  __bf16* wvbf = (__bf16*)carve(wN * 2);
  __bf16* Qh   = (__bf16*)carve(actN * 2);
  __bf16* Kh   = (__bf16*)carve(actN * 2);
  __bf16* Vt   = (__bf16*)carve(actN * 2);
  float*  Yb   = (float*)carve(actN * 4);

  {
    int n8 = (int)(actN / 8);
    mha_cvt_bf16<<<n8 / 256, 256, 0, stream>>>(q, qbf, n8);
    mha_cvt_bf16<<<n8 / 256, 256, 0, stream>>>(k, kbf, n8);
    mha_cvt_bf16<<<n8 / 256, 256, 0, stream>>>(v, vbf, n8);
  }
  {
    int n8 = (int)(wN / 8);
    mha_cvt_bf16<<<n8 / 256, 256, 0, stream>>>(Wq, wqbf, n8);
    mha_cvt_bf16<<<n8 / 256, 256, 0, stream>>>(Wk, wkbf, n8);
    mha_cvt_bf16<<<n8 / 256, 256, 0, stream>>>(Wv, wvbf, n8);
  }
  dim3 pg(H_, NTOK_ / 16);                  // 8 x 512 waves, 16x64 tile each
  mha_qkv_proj<<<pg, 32, 0, stream>>>(qbf, wqbf, bq, Qh, 0);
  mha_qkv_proj<<<pg, 32, 0, stream>>>(kbf, wkbf, bk, Kh, 0);
  mha_qkv_proj<<<pg, 32, 0, stream>>>(vbf, wvbf, bv, Vt, 1);

  dim3 ag(S_ / 32, B_ * H_);                // 32 x 64 waves, 32 queries each
  mha_attention<<<ag, 32, 0, stream>>>(Qh, Kh, Vt, qmask, kmask, Yb);

  mha_ln<<<NTOK_, 32, 0, stream>>>(Yb, q, gamma, beta, out);
}
